// MultiHeadAttention_37726992728436
// MI455X (gfx1250) — matrix operand order repaired and checked by reading, not compiled, not run on hardware
//
#include <hip/hip_runtime.h>
#include <math.h>

// ---------------------------------------------------------------------------
// Problem constants (B=4, T=2048, E=1024, H=16, D=64)
// ---------------------------------------------------------------------------
#define B_ 4
#define T_ 2048
#define E_ 1024
#define H_ 16
#define D_ 64

#define SROWS 16     // rows per fused scores+softmax block
#define LDP   2052   // padded LDS row pitch (floats) to break bank alignment

typedef __attribute__((ext_vector_type(16))) __bf16 v16bf;
typedef __attribute__((ext_vector_type(8)))  float  v8f;

union ABFrag {
    uint4 u2[2];
    v16bf v;
};

// float -> bf16 (round to nearest even, ignoring NaN edge case)
static __device__ __forceinline__ unsigned short f2bf(float f) {
    unsigned int u = __float_as_uint(f);
    u += 0x7FFFu + ((u >> 16) & 1u);
    return (unsigned short)(u >> 16);
}

static __device__ __forceinline__ v8f wmma_bf16(v16bf a, v16bf b, v8f c) {
    // (neg_a, A, neg_b, B, c_mod, C, reuse_a, reuse_b)
    return __builtin_amdgcn_wmma_f32_16x16x32_bf16(false, a, false, b,
                                                   (short)0, c, false, false);
}

// A-fragment (16x32 bf16, row-major source, stride ld in elements).
// Lane L<16: M=L, K={0..7,16..23}; lane>=16: M=L-16, K={8..15,24..31}.
static __device__ __forceinline__ v16bf load_a_bf16(const unsigned short* base, int ld) {
    int lane = threadIdx.x & 31;
    const unsigned short* p = base + (size_t)(lane & 15) * ld + ((lane >> 4) << 3);
    ABFrag f;
    f.u2[0] = *(const uint4*)(p);        // K chunk 0 (8 halves)
    f.u2[1] = *(const uint4*)(p + 16);   // K chunk 1 (8 halves)
    return f.v;
}

// B-fragment (32x16 bf16). Source matrix is [N, K] row-major (stride ld), so
// output column n = row n of source (contiguous K).
// Lane L<16: N=L, K={0..7,16..23}; lane>=16: N=L-16, K={8..15,24..31}.
static __device__ __forceinline__ v16bf load_b_bf16(const unsigned short* base, int ld) {
    int lane = threadIdx.x & 31;
    const unsigned short* p = base + (size_t)(lane & 15) * ld + ((lane >> 4) << 3);
    ABFrag f;
    f.u2[0] = *(const uint4*)(p);
    f.u2[1] = *(const uint4*)(p + 16);
    return f.v;
}

// A-fragment from fp32 source (convert to bf16 in registers).
static __device__ __forceinline__ v16bf load_a_f32(const float* base, int ld) {
    int lane = threadIdx.x & 31;
    const float* p = base + (size_t)(lane & 15) * ld + ((lane >> 4) << 3);
    float4 a = *(const float4*)(p);
    float4 b = *(const float4*)(p + 4);
    float4 c = *(const float4*)(p + 16);
    float4 d = *(const float4*)(p + 20);
    ABFrag f;
    f.u2[0].x = (unsigned)f2bf(a.x) | ((unsigned)f2bf(a.y) << 16);
    f.u2[0].y = (unsigned)f2bf(a.z) | ((unsigned)f2bf(a.w) << 16);
    f.u2[0].z = (unsigned)f2bf(b.x) | ((unsigned)f2bf(b.y) << 16);
    f.u2[0].w = (unsigned)f2bf(b.z) | ((unsigned)f2bf(b.w) << 16);
    f.u2[1].x = (unsigned)f2bf(c.x) | ((unsigned)f2bf(c.y) << 16);
    f.u2[1].y = (unsigned)f2bf(c.z) | ((unsigned)f2bf(c.w) << 16);
    f.u2[1].z = (unsigned)f2bf(d.x) | ((unsigned)f2bf(d.y) << 16);
    f.u2[1].w = (unsigned)f2bf(d.z) | ((unsigned)f2bf(d.w) << 16);
    return f.v;
}

// ---------------------------------------------------------------------------
// 1) fp32 -> bf16 cast (vectorized by 4)
// ---------------------------------------------------------------------------
__global__ __launch_bounds__(256) void cast_bf16_kernel(const float* __restrict__ src,
                                                        unsigned short* __restrict__ dst,
                                                        int n4) {
    int i = blockIdx.x * 256 + threadIdx.x;
    if (i >= n4) return;
    float4 f = ((const float4*)src)[i];
    ushort4 o;
    o.x = f2bf(f.x); o.y = f2bf(f.y); o.z = f2bf(f.z); o.w = f2bf(f.w);
    ((ushort4*)dst)[i] = o;
}

// ---------------------------------------------------------------------------
// 2) QKV projection: Y[8192,3072] = X[8192,1024] @ W^T + bias
//    n in [0,1024):    Q (scaled by D^-0.5) -> qb  [B,T,E] bf16
//    n in [1024,2048): K                    -> kb  [B,T,E] bf16
//    n in [2048,3072): V transposed         -> vt  [B*H, D, T] bf16
// Block = 256 threads = 8 waves (2 M x 4 N), block tile 64x128.
// ---------------------------------------------------------------------------
__global__ __launch_bounds__(256) void qkv_gemm_kernel(const unsigned short* __restrict__ xb,
                                                       const unsigned short* __restrict__ wb,
                                                       const float* __restrict__ bias,
                                                       unsigned short* __restrict__ qb,
                                                       unsigned short* __restrict__ kb,
                                                       unsigned short* __restrict__ vt) {
    int wave = threadIdx.x >> 5;
    int wm = wave >> 2, wn = wave & 3;
    int m0 = blockIdx.y * 64 + wm * 32;
    int n0 = blockIdx.x * 128 + wn * 32;

    v8f acc[2][2] = {};
    for (int k = 0; k < E_; k += 32) {
        __builtin_prefetch(xb + (size_t)m0 * E_ + k + 64, 0, 1);
        __builtin_prefetch(wb + (size_t)n0 * E_ + k + 64, 0, 1);
        v16bf a0 = load_a_bf16(xb + (size_t)m0 * E_ + k, E_);
        v16bf a1 = load_a_bf16(xb + (size_t)(m0 + 16) * E_ + k, E_);
        v16bf b0 = load_b_bf16(wb + (size_t)n0 * E_ + k, E_);
        v16bf b1 = load_b_bf16(wb + (size_t)(n0 + 16) * E_ + k, E_);
        acc[0][0] = wmma_bf16(a0, b0, acc[0][0]);
        acc[0][1] = wmma_bf16(a0, b1, acc[0][1]);
        acc[1][0] = wmma_bf16(a1, b0, acc[1][0]);
        acc[1][1] = wmma_bf16(a1, b1, acc[1][1]);
    }

    int lane = threadIdx.x & 31;
    int colL = lane & 15;
    int rowH = (lane >> 4) << 3;
    const float scale = 0.125f;  // D^-0.5

    for (int i = 0; i < 2; ++i)
        for (int j = 0; j < 2; ++j) {
            int n = n0 + j * 16 + colL;
            float bv = bias[n];
            for (int r = 0; r < 8; ++r) {
                int m = m0 + i * 16 + rowH + r;          // global row in [0, B*T)
                float val = acc[i][j][r] + bv;
                if (n < E_) {
                    qb[(size_t)m * E_ + n] = f2bf(val * scale);
                } else if (n < 2 * E_) {
                    kb[(size_t)m * E_ + (n - E_)] = f2bf(val);
                } else {
                    int nn = n - 2 * E_;                 // h*64 + d
                    int h = nn >> 6, d = nn & 63;
                    int b = m >> 11, t = m & (T_ - 1);
                    vt[((size_t)(b * H_ + h) * D_ + d) * T_ + t] = f2bf(val);
                }
            }
        }
}

// ---------------------------------------------------------------------------
// 3) Fused scores + softmax, per (b,h):
//    Each block computes 16 complete rows of S = Qh @ Kh^T (2048 cols) into
//    LDS, then does the row softmax there and writes probs/scale (= probs*8)
//    to global exactly once.  LDS use: 16 * 2052 * 4 B = 128.25 KiB.
//    8 waves: wave w computes cols [w*256, w*256+256); A-fragments for the 16
//    rows live in registers for the whole kernel.  Softmax: wave w reduces
//    rows 2w, 2w+1 entirely within the wave (shfl only).
// ---------------------------------------------------------------------------
__global__ __launch_bounds__(256) void scores_softmax_kernel(
        const unsigned short* __restrict__ qb,
        const unsigned short* __restrict__ kb,
        float* __restrict__ probs) {
    extern __shared__ float sm[];     // [SROWS][LDP]
    int bh = blockIdx.z;
    int b = bh >> 4, h = bh & 15;
    int wave = threadIdx.x >> 5;
    int lane = threadIdx.x & 31;
    int m0 = blockIdx.y * SROWS;

    const unsigned short* qh = qb + (size_t)b * T_ * E_ + h * D_;
    const unsigned short* kh = kb + (size_t)b * T_ * E_ + h * D_;

    // A fragments (16 rows x K=64) held in registers: k-steps 0 and 32
    v16bf a0 = load_a_bf16(qh + (size_t)m0 * E_ + 0, E_);
    v16bf a1 = load_a_bf16(qh + (size_t)m0 * E_ + 32, E_);

    int colL = lane & 15;
    int rowH = (lane >> 4) << 3;

    for (int nc = 0; nc < 8; ++nc) {
        int n0 = wave * 256 + nc * 32;
        v16bf b00 = load_b_bf16(kh + (size_t)n0 * E_ + 0, E_);
        v16bf b01 = load_b_bf16(kh + (size_t)n0 * E_ + 32, E_);
        v16bf b10 = load_b_bf16(kh + (size_t)(n0 + 16) * E_ + 0, E_);
        v16bf b11 = load_b_bf16(kh + (size_t)(n0 + 16) * E_ + 32, E_);
        v8f acc0 = {}, acc1 = {};
        acc0 = wmma_bf16(a0, b00, acc0);
        acc0 = wmma_bf16(a1, b01, acc0);
        acc1 = wmma_bf16(a0, b10, acc1);
        acc1 = wmma_bf16(a1, b11, acc1);
#pragma unroll
        for (int r = 0; r < 8; ++r) {
            int row = rowH + r;
            sm[row * LDP + n0 + colL]      = acc0[r];
            sm[row * LDP + n0 + 16 + colL] = acc1[r];
        }
    }
    __syncthreads();

    // softmax: wave handles rows 2*wave and 2*wave+1; 64 values per lane
    for (int rr = 0; rr < 2; ++rr) {
        int row = wave * 2 + rr;
        const float* rp = sm + row * LDP;
        float v[64];
#pragma unroll
        for (int i = 0; i < 16; ++i) {
            float4 q4 = *(const float4*)(rp + i * 128 + lane * 4);
            v[i * 4 + 0] = q4.x; v[i * 4 + 1] = q4.y;
            v[i * 4 + 2] = q4.z; v[i * 4 + 3] = q4.w;
        }
        float mx = v[0];
#pragma unroll
        for (int i = 1; i < 64; ++i) mx = fmaxf(mx, v[i]);
#pragma unroll
        for (int off = 16; off > 0; off >>= 1) mx = fmaxf(mx, __shfl_xor(mx, off, 32));
        float s = 0.0f;
#pragma unroll
        for (int i = 0; i < 64; ++i) { v[i] = __expf(v[i] - mx); s += v[i]; }
#pragma unroll
        for (int off = 16; off > 0; off >>= 1) s += __shfl_xor(s, off, 32);
        float f = 8.0f / s;  // probs / scale
        float* gp = probs + (size_t)(bh * T_ + m0 + row) * T_;
#pragma unroll
        for (int i = 0; i < 16; ++i) {
            float4 o;
            o.x = v[i * 4 + 0] * f; o.y = v[i * 4 + 1] * f;
            o.z = v[i * 4 + 2] * f; o.w = v[i * 4 + 3] * f;
            *(float4*)(gp + i * 128 + lane * 4) = o;
        }
    }
}

// ---------------------------------------------------------------------------
// 4) attn = (probs*8) @ V * 0.125  -> ctx [B,T,E] bf16
//    Per (b,h): M=2048, N=64, Kdim=2048.
//    8 waves stacked in M (block tile 256x64); each wave covers the full N
//    (2 M-tiles x 4 N-tiles) so every probs element is loaded exactly once
//    per block; the tiny V tile (64x2048 bf16 = 256 KB) stays cache-resident.
// ---------------------------------------------------------------------------
__global__ __launch_bounds__(256) void attn_kernel(const float* __restrict__ probs,
                                                   const unsigned short* __restrict__ vt,
                                                   unsigned short* __restrict__ ctx) {
    int bh = blockIdx.z;
    int b = bh >> 4, h = bh & 15;
    int wave = threadIdx.x >> 5;
    int m0 = blockIdx.y * 256 + wave * 32;

    const float* ph = probs + (size_t)bh * T_ * T_;
    const unsigned short* vh = vt + (size_t)bh * D_ * T_;

    v8f acc[2][4] = {};
    for (int k = 0; k < T_; k += 32) {
        __builtin_prefetch(ph + (size_t)m0 * T_ + k + 64, 0, 1);
        v16bf a0 = load_a_f32(ph + (size_t)m0 * T_ + k, T_);
        v16bf a1 = load_a_f32(ph + (size_t)(m0 + 16) * T_ + k, T_);
#pragma unroll
        for (int j = 0; j < 4; ++j) {
            v16bf bj = load_b_bf16(vh + (size_t)(j * 16) * T_ + k, T_);
            acc[0][j] = wmma_bf16(a0, bj, acc[0][j]);
            acc[1][j] = wmma_bf16(a1, bj, acc[1][j]);
        }
    }

    int lane = threadIdx.x & 31;
    int colL = lane & 15;
    int rowH = (lane >> 4) << 3;
    for (int i = 0; i < 2; ++i)
        for (int j = 0; j < 4; ++j)
            for (int r = 0; r < 8; ++r) {
                int t = m0 + i * 16 + rowH + r;
                int d = j * 16 + colL;
                ctx[(size_t)(b * T_ + t) * E_ + h * D_ + d] = f2bf(acc[i][j][r] * 0.125f);
            }
}

// ---------------------------------------------------------------------------
// 5) Output projection: out[8192,1024] = ctx @ out_w^T + out_b (fp32)
// ---------------------------------------------------------------------------
__global__ __launch_bounds__(256) void outproj_kernel(const unsigned short* __restrict__ ctx,
                                                      const unsigned short* __restrict__ owb,
                                                      const float* __restrict__ ob,
                                                      float* __restrict__ out) {
    int wave = threadIdx.x >> 5;
    int wm = wave >> 2, wn = wave & 3;
    int m0 = blockIdx.y * 64 + wm * 32;
    int n0 = blockIdx.x * 128 + wn * 32;

    v8f acc[2][2] = {};
    for (int k = 0; k < E_; k += 32) {
        __builtin_prefetch(ctx + (size_t)m0 * E_ + k + 64, 0, 1);
        __builtin_prefetch(owb + (size_t)n0 * E_ + k + 64, 0, 1);
        v16bf a0 = load_a_bf16(ctx + (size_t)m0 * E_ + k, E_);
        v16bf a1 = load_a_bf16(ctx + (size_t)(m0 + 16) * E_ + k, E_);
        v16bf b0 = load_b_bf16(owb + (size_t)n0 * E_ + k, E_);
        v16bf b1 = load_b_bf16(owb + (size_t)(n0 + 16) * E_ + k, E_);
        acc[0][0] = wmma_bf16(a0, b0, acc[0][0]);
        acc[0][1] = wmma_bf16(a0, b1, acc[0][1]);
        acc[1][0] = wmma_bf16(a1, b0, acc[1][0]);
        acc[1][1] = wmma_bf16(a1, b1, acc[1][1]);
    }

    int lane = threadIdx.x & 31;
    int colL = lane & 15;
    int rowH = (lane >> 4) << 3;
    for (int i = 0; i < 2; ++i)
        for (int j = 0; j < 2; ++j) {
            int n = n0 + j * 16 + colL;
            float bv = ob[n];
            for (int r = 0; r < 8; ++r) {
                int m = m0 + i * 16 + rowH + r;
                out[(size_t)m * E_ + n] = acc[i][j][r] + bv;
            }
        }
}

// ---------------------------------------------------------------------------
// Host launcher
// ---------------------------------------------------------------------------
extern "C" void kernel_launch(void* const* d_in, const int* in_sizes, int n_in,
                              void* d_out, int out_size, void* d_ws, size_t ws_size,
                              hipStream_t stream) {
    const float* x    = (const float*)d_in[0];  // [B,T,E]
    const float* wi   = (const float*)d_in[1];  // [3E,E]
    const float* bi   = (const float*)d_in[2];  // [3E]
    const float* ow   = (const float*)d_in[3];  // [E,E]
    const float* ob   = (const float*)d_in[4];  // [E]

    float* out   = (float*)d_out;                       // [B,T,E]
    float* probs = out + (size_t)B_ * T_ * E_;          // [B,H,T,T] (probs/scale)

    char* ws = (char*)d_ws;
    unsigned short* xb  = (unsigned short*)(ws);                        // 16 MiB
    unsigned short* wb  = (unsigned short*)(ws + (16ull << 20));        //  6 MiB
    unsigned short* owb = (unsigned short*)(ws + (22ull << 20));        //  2 MiB
    unsigned short* qb  = (unsigned short*)(ws + (24ull << 20));        // 16 MiB
    unsigned short* kb  = (unsigned short*)(ws + (40ull << 20));        // 16 MiB
    unsigned short* vt  = (unsigned short*)(ws + (56ull << 20));        // 16 MiB
    unsigned short* ctx = (unsigned short*)(ws + (72ull << 20));        // 16 MiB
    // total workspace use: 88 MiB

    // 1) casts
    cast_bf16_kernel<<<dim3(8192), 256, 0, stream>>>(x,  xb,  (B_ * T_ * E_) / 4);
    cast_bf16_kernel<<<dim3(3072), 256, 0, stream>>>(wi, wb,  (3 * E_ * E_) / 4);
    cast_bf16_kernel<<<dim3(1024), 256, 0, stream>>>(ow, owb, (E_ * E_) / 4);

    // 2) QKV projection (M=8192, N=3072)
    qkv_gemm_kernel<<<dim3(3072 / 128, 8192 / 64), 256, 0, stream>>>(xb, wb, bi, qb, kb, vt);

    // 3) fused scores + softmax per (b,h): 16 full rows per block
    size_t lds_bytes = (size_t)SROWS * LDP * sizeof(float);  // 128.25 KiB
    scores_softmax_kernel<<<dim3(1, T_ / SROWS, B_ * H_), 256, lds_bytes, stream>>>(qb, kb, probs);

    // 4) attention context (M=2048 per head, block tile 256x64)
    attn_kernel<<<dim3(1, T_ / 256, B_ * H_), 256, 0, stream>>>(probs, vt, ctx);

    // 5) output projection (M=8192, N=1024)
    outproj_kernel<<<dim3(1024 / 128, 8192 / 64), 256, 0, stream>>>(ctx, owb, ob, out);
}
